// SimpleSSM_15169824489977
// MI455X (gfx1250) — matrix operand order repaired and checked by reading, not compiled, not run on hardware
//
#include <hip/hip_runtime.h>
#include <hip/hip_bf16.h>

// Problem constants (match reference)
#define BB   8
#define TT   4096
#define CC   512
#define SS   128                 // time-chunk length
#define NCH  (TT / SS)           // 32 chunks
#define LDSP 520                 // LDS row pitch in halves: 520*2B=1040B, 16B aligned, bank-stride 4

typedef _Float16 v16h __attribute__((ext_vector_type(16)));
typedef _Float16 v8h  __attribute__((ext_vector_type(8)));
typedef float    v8f  __attribute__((ext_vector_type(8)));

__device__ __forceinline__ void decay_params(const float* __restrict__ logd, int c,
                                             float& d, float& lnd) {
    float z = logd[c];
    // numerically-stable sigmoid, clamped away from exact 0 to avoid 0*(-inf)
    float e;
    if (z >= 0.f) { e = __builtin_expf(-z); d = 1.f / (1.f + e); }
    else          { e = __builtin_expf(z);  d = e / (1.f + e);   }
    d   = fmaxf(d, 1e-30f);
    lnd = __builtin_logf(d);     // <= 0
}

// ---------------------------------------------------------------------------
// K0: repack mix_w (fp32 [n][k]) -> f16 WMMA-B fragment order:
//     Wp[((kt*512 + n)*2 + khalf)*16 + kk] = W[n][kt*32 + (kk>>3)*16 + khalf*8 + (kk&7)]
// so a B fragment (n-tile, k-tile) is 32 contiguous bytes per lane and the
// wave's 32 lane-loads tile 1KB contiguously.
// ---------------------------------------------------------------------------
__global__ __launch_bounds__(256) void k_pack_w(const float* __restrict__ W,
                                                _Float16* __restrict__ Wp) {
    int o  = blockIdx.x * 256 + threadIdx.x;       // 0 .. 512*512-1
    int kk = o & 15;
    int kh = (o >> 4) & 1;
    int n  = (o >> 5) & (CC - 1);
    int kt = o >> 14;                              // 0..15
    int k  = kt * 32 + ((kk >> 3) << 4) + kh * 8 + (kk & 7);
    Wp[o]  = (_Float16)W[n * CC + k];
}

// ---------------------------------------------------------------------------
// K1: per (b, chunk): partial[c] = sum_{s in chunk} d^s * x[b,s,c]
// Lanes map to channels -> fully coalesced 128B lines per step.
// ---------------------------------------------------------------------------
__global__ __launch_bounds__(256) void k_partials(const float* __restrict__ x,
                                                  const float* __restrict__ logd,
                                                  float* __restrict__ part) {
    int b  = blockIdx.x / NCH;
    int ch = blockIdx.x % NCH;
    int s0 = ch * SS;
    for (int i = 0; i < 2; ++i) {
        int c = threadIdx.x + (i << 8);
        float d, lnd;
        decay_params(logd, c, d, lnd);
        float p   = __builtin_expf((float)s0 * lnd);   // d^{s0}
        float acc = 0.f;
        const float* xp = x + ((size_t)b * TT + s0) * CC + c;
        #pragma unroll 4
        for (int s = 0; s < SS; ++s) {
            acc += p * xp[(size_t)s * CC];
            p *= d;
        }
        part[((size_t)b * NCH + ch) * CC + c] = acc;
    }
}

// ---------------------------------------------------------------------------
// K2: exclusive prefix over the 32 chunks, in place, one thread per (b,c)
// ---------------------------------------------------------------------------
__global__ __launch_bounds__(256) void k_prefix(float* __restrict__ part) {
    int idx = blockIdx.x * 256 + threadIdx.x;      // 0 .. B*C-1
    int b = idx >> 9, c = idx & (CC - 1);
    float run = 0.f;
    for (int ch = 0; ch < NCH; ++ch) {
        size_t o = ((size_t)b * NCH + ch) * CC + c;
        float v  = part[o];
        part[o]  = run;
        run     += v;
    }
}

// ---------------------------------------------------------------------------
// K3 (fused): finish the scan for a 128-row tile -> f16 y in LDS, then
// out[128 x 512] = y @ W^T + bias via v_wmma_f32_16x16x32_f16.
// Blocking: wave w owns n-tiles [4w, 4w+4); 8 m-tile accumulators stay live,
// each B fragment is loaded once per block (W read exactly once per block
// from L2) and is software-pipelined one kt ahead of the 8 WMMAs.
// kt/ntl loops are kept rolled (#pragma unroll 1) to bound VGPR pressure
// (~120 live VGPRs) and avoid the scratch spills seen with full unrolling.
// ---------------------------------------------------------------------------
__global__ __launch_bounds__(256) void k_scan_mix(const float* __restrict__ x,
                                                  const float* __restrict__ logd,
                                                  const float* __restrict__ part,
                                                  const _Float16* __restrict__ Wp,
                                                  const float* __restrict__ bias,
                                                  float* __restrict__ out) {
    extern __shared__ _Float16 yl[];               // [128][LDSP] = 130 KB
    int b  = blockIdx.x / NCH;
    int ch = blockIdx.x % NCH;
    int s0 = ch * SS;

    // ---- Phase A: scan 128 steps for 2 channels per thread, y -> LDS (f16)
    for (int i = 0; i < 2; ++i) {
        int c = threadIdx.x + (i << 8);
        float d, lnd;
        decay_params(logd, c, d, lnd);
        float p   = __builtin_expf((float)s0 * lnd);                 // d^{s0+s}
        float run = part[((size_t)b * NCH + ch) * CC + c];           // exclusive prefix
        const float* xp = x + ((size_t)b * TT + s0) * CC + c;
        for (int s = 0; s < SS; ++s) {
            run += p * xp[(size_t)s * CC];
            p *= d;
            float q = __builtin_expf((float)(TT - 1 - (s0 + s)) * lnd); // d^{T-1-t}
            yl[s * LDSP + c] = (_Float16)(q * run);
        }
    }
    __syncthreads();

    // ---- Phase B: WMMA GEMM
    int lane = threadIdx.x & 31;
    int wave = threadIdx.x >> 5;
    int m    = lane & 15;                          // row within m-tile / col within n-tile
    int kh   = lane >> 4;                          // lane half
    size_t gRow0 = (size_t)b * TT + s0;            // global row of tile row 0

    // per-lane bases (column part added per ntl)
    const _Float16* yBase = &yl[m * LDSP + kh * 8];          // + mt*16*LDSP + kt*32
    float* oBase = out + (gRow0 + (kh << 3)) * CC + m;       // + (mt*16+r)*CC + n0

    #pragma unroll 1
    for (int ntl = 0; ntl < 4; ++ntl) {
        int n0 = (wave * 4 + ntl) * 16;            // this wave's output column tile
        float bv = bias[n0 + m];
        v8f acc[8];
        #pragma unroll
        for (int mt = 0; mt < 8; ++mt)
            #pragma unroll
            for (int r = 0; r < 8; ++r)
                acc[mt][r] = bv;

        // B fragment stream: lane = (column n0+m, half kh) -> 32 contiguous bytes
        const _Float16* bCol = Wp + (((size_t)(n0 + m)) * 2 + kh) * 16;
        v16h bm = *(const v16h*)bCol;              // kt = 0

        #pragma unroll 1
        for (int kt = 0; kt < 16; ++kt) {
            // prefetch next B fragment (clamped, branch-free) to hide L2 latency
            int ktn = kt + 1 < 16 ? kt + 1 : 15;
            v16h bnext = *(const v16h*)(bCol + (size_t)ktn * (CC * 32));
            #pragma unroll
            for (int mt = 0; mt < 8; ++mt) {
                // A fragment (ISA 16-bit A 16x32 layout):
                //   lanes<16: K in [0,8) u [16,24); lanes>=16: K in [8,16) u [24,32)
                const _Float16* ap = yBase + mt * (16 * LDSP) + kt * 32;
                v8h alo = *(const v8h*)ap;         // 16B-aligned LDS loads
                v8h ahi = *(const v8h*)(ap + 16);
                v16h a  = __builtin_shufflevector(alo, ahi,
                            0, 1, 2, 3, 4, 5, 6, 7, 8, 9, 10, 11, 12, 13, 14, 15);
                acc[mt] = __builtin_amdgcn_wmma_f32_16x16x32_f16(
                              false, a, false, bm, (short)0, acc[mt], false, false);
            }
            bm = bnext;
        }

        // D layout: VGPR r -> row r (lanes 0-15) / row r+8 (lanes 16-31), col = n0+m
        float* op = oBase + n0;
        #pragma unroll
        for (int mt = 0; mt < 8; ++mt)
            #pragma unroll
            for (int r = 0; r < 8; ++r)
                op[(size_t)(mt * 16 + r) * CC] = acc[mt][r];
    }
}

// ---------------------------------------------------------------------------
extern "C" void kernel_launch(void* const* d_in, const int* in_sizes, int n_in,
                              void* d_out, int out_size, void* d_ws, size_t ws_size,
                              hipStream_t stream) {
    const float* x    = (const float*)d_in[0];     // [B,T,C]
    const float* logd = (const float*)d_in[1];     // [C]
    const float* W    = (const float*)d_in[2];     // [C,C]
    const float* bias = (const float*)d_in[3];     // [C]
    float* out = (float*)d_out;                    // [B,T,C]

    // workspace layout: [0, 512KB) chunk partials fp32 ; [512KB, 1MB) packed W f16
    float*    part = (float*)d_ws;
    _Float16* Wp   = (_Float16*)((char*)d_ws + (size_t)BB * NCH * CC * sizeof(float));

    k_pack_w  <<<(CC * CC) / 256, 256, 0, stream>>>(W, Wp);
    k_partials<<<BB * NCH,        256, 0, stream>>>(x, logd, part);
    k_prefix  <<<(BB * CC) / 256, 256, 0, stream>>>(part);
    k_scan_mix<<<BB * NCH, 256, SS * LDSP * sizeof(_Float16), stream>>>(
        x, logd, part, Wp, bias, out);
}
